// ScaledDotProductAttention_111669149958
// MI455X (gfx1250) — compile-verified, matrix operand order checked by reading, not run on hardware
//
#include <hip/hip_runtime.h>

// Causal scaled-dot-product attention for MI455X (gfx1250, wave32, WMMA).
// B=2, H=16, S=2048, DK=64. Outputs: context [B,H,S,DK] then attn [B,H,S,S].

#define Bc  2
#define Hc  16
#define Sc  2048
#define DKc 64

typedef __attribute__((ext_vector_type(16))) __bf16 v16bf;
typedef __attribute__((ext_vector_type(8)))  float  v8f;
typedef __attribute__((ext_vector_type(4)))  float  v4f;

// K-index pattern for 16-bit A/B fragments (16x32, ISA 7.12.2):
// low lanes hold K {0..7,16..23}, high lanes K {8..15,24..31}, 2 per VGPR.
__device__ __forceinline__ int kpat(int j, int hi) {
  int base = (j < 4) ? (2 * j) : (16 + 2 * (j - 4));
  return base + (hi ? 8 : 0);
}

__device__ __forceinline__ v8f wmma_acc(v16bf a, v16bf b, v8f c) {
  return __builtin_amdgcn_wmma_f32_16x16x32_bf16(false, a, false, b, (short)0, c,
                                                 false, false);
}

// ds_swizzle XOR within 16-lane halves: offset = xor<<10 | and_mask(0x1f)
template <int IMM>
__device__ __forceinline__ float swzf(float x) {
  return __int_as_float(__builtin_amdgcn_ds_swizzle(__float_as_int(x), IMM));
}
__device__ __forceinline__ float redmax16(float x) {
  x = fmaxf(x, swzf<0x041f>(x));   // xor 1
  x = fmaxf(x, swzf<0x081f>(x));   // xor 2
  x = fmaxf(x, swzf<0x101f>(x));   // xor 4
  x = fmaxf(x, swzf<0x201f>(x));   // xor 8
  return x;
}
__device__ __forceinline__ float redsum16(float x) {
  x += swzf<0x041f>(x);
  x += swzf<0x081f>(x);
  x += swzf<0x101f>(x);
  x += swzf<0x201f>(x);
  return x;
}

// A-fragment of Q (pre-scaled by 1/sqrt(dk); exact in bf16 since 2^-3).
__device__ __forceinline__ v16bf load_q_frag(const float* __restrict__ qrow, int hi, int c) {
  v16bf a;
#pragma unroll
  for (int e = 0; e < 16; ++e) {
    int k = kpat(e >> 1, hi) + (e & 1) + 32 * c;
    a[e] = (__bf16)(qrow[k] * 0.125f);
  }
  return a;
}

// B-fragment of K^T from LDS (column n = lane&15 is key row ksub*16+n)
__device__ __forceinline__ v16bf load_k_frag(const __bf16 (*klds)[68], int ksub,
                                             int lane, int c) {
  int key = ksub * 16 + (lane & 15);
  int hi  = lane >> 4;
  v16bf b;
#pragma unroll
  for (int e = 0; e < 16; ++e) {
    int k = kpat(e >> 1, hi) + (e & 1) + 32 * c;
    b[e] = klds[key][k];
  }
  return b;
}

// Pack 4 floats to 4 bf16 and store as one 8-byte LDS write (dst 8B-aligned).
__device__ __forceinline__ void store_bf16x4(__bf16* dst, float4 v) {
  __bf16 h[4] = {(__bf16)v.x, (__bf16)v.y, (__bf16)v.z, (__bf16)v.w};
  *(uint2*)dst = *(const uint2*)h;
}

__global__ __launch_bounds__(256, 2) void fa_fwd_kernel(
    const float* __restrict__ Q, const float* __restrict__ K,
    const float* __restrict__ V, float* __restrict__ ctx_out,
    float* __restrict__ attn_out) {
  __shared__ __bf16 Klds[2][32][68];     // double-buffered 32 keys x 64 dims
  __shared__ __bf16 Vtlds[2][64][36];    // double-buffered transposed [dim][key]
  __shared__ __bf16 Plds[8][16][34];     // per-wave P-tile staging (padded)

  const int tid  = threadIdx.x;
  const int lane = tid & 31;
  const int wv   = tid >> 5;
  const int hi   = lane >> 4;
  const int ln   = lane & 15;

  const int q0  = blockIdx.x * 128;      // 128 queries per workgroup
  const int bh  = blockIdx.y;            // fused batch*head
  const int q0w = q0 + wv * 16;          // 16 queries per wave

  const float* Qb = Q + (size_t)bh * Sc * DKc;
  const float* Kb = K + (size_t)bh * Sc * DKc;
  const float* Vb = V + (size_t)bh * Sc * DKc;
  float* ctxb  = ctx_out  + (size_t)bh * Sc * DKc;
  float* attnb = attn_out + (size_t)bh * Sc * Sc;

  // Per-wave Q A-fragments (loaded once, pre-scaled)
  const float* qrow = Qb + (size_t)(q0w + ln) * DKc;
  const v16bf qa0 = load_q_frag(qrow, hi, 0);
  const v16bf qa1 = load_q_frag(qrow, hi, 1);

  const int qmax_w  = q0w + 15;
  const int kbNeed  = qmax_w >> 5;         // last (diagonal) 32-key block
  const int kbCount = (q0 >> 5) + 4;       // key blocks for whole workgroup

  const int e0 = tid * 4;                  // staging element ids (float4)
  const int e1 = tid * 4 + 1024;
  const int key0 = e0 >> 6, dim0 = e0 & 63;
  const int key1 = e1 >> 6, dim1 = e1 & 63;

  // Per-LANE online softmax stats over this lane's columns only.
  float ml[8], sl[8];
#pragma unroll
  for (int r = 0; r < 8; ++r) { ml[r] = -3.0e38f; sl[r] = 0.0f; }

  // ---------------- Pass 1: per-lane row max & sum -------------------------
  {
    store_bf16x4(&Klds[0][key0][dim0], *(const float4*)(Kb + e0));
    store_bf16x4(&Klds[0][key1][dim1], *(const float4*)(Kb + e1));
    __syncthreads();

    for (int kbi = 0; kbi < kbCount; ++kbi) {
      const bool haveNext = (kbi + 1 < kbCount);
      float4 kr0, kr1;
      if (haveNext) {                      // issue next block's loads early
        const float* src = Kb + (size_t)(kbi + 1) * 32 * DKc;
        kr0 = *(const float4*)(src + e0);
        kr1 = *(const float4*)(src + e1);
      }
      if (kbi <= kbNeed) {                 // wave-uniform
        const __bf16 (*kcur)[68] = Klds[kbi & 1];
        v8f s0 = {}, s1 = {};
        {
          v16bf b = load_k_frag(kcur, 0, lane, 0);
          s0 = wmma_acc(qa0, b, s0);
          b = load_k_frag(kcur, 0, lane, 1);
          s0 = wmma_acc(qa1, b, s0);
          b = load_k_frag(kcur, 1, lane, 0);
          s1 = wmma_acc(qa0, b, s1);
          b = load_k_frag(kcur, 1, lane, 1);
          s1 = wmma_acc(qa1, b, s1);
        }
        if (kbi == kbNeed) {               // diagonal block: apply causal mask
          const int kb = kbi * 32;
          const int col0 = kb + ln, col1 = kb + 16 + ln;
          const int rowBase = q0w + (hi ? 8 : 0);
#pragma unroll
          for (int r = 0; r < 8; ++r) {
            const int row = rowBase + r;
            float sv0 = (col0 > row) ? -1.0e9f : s0[r];
            float sv1 = (col1 > row) ? -1.0e9f : s1[r];
            float mn = fmaxf(ml[r], fmaxf(sv0, sv1));
            float corr = __expf(ml[r] - mn);
            sl[r] = sl[r] * corr + __expf(sv0 - mn) + __expf(sv1 - mn);
            ml[r] = mn;
          }
        } else {                           // interior block: no mask needed
#pragma unroll
          for (int r = 0; r < 8; ++r) {
            float sv0 = s0[r], sv1 = s1[r];
            float mn = fmaxf(ml[r], fmaxf(sv0, sv1));
            float corr = __expf(ml[r] - mn);
            sl[r] = sl[r] * corr + __expf(sv0 - mn) + __expf(sv1 - mn);
            ml[r] = mn;
          }
        }
      }
      if (haveNext) {                      // convert + store into other buffer
        __bf16 (*knxt)[68] = Klds[(kbi + 1) & 1];
        store_bf16x4(&knxt[key0][dim0], kr0);
        store_bf16x4(&knxt[key1][dim1], kr1);
      }
      __syncthreads();
    }
  }

  // Cross-lane combine (once): lo = m_row + ln(l_row); p = exp(s - lo).
  float lo[8];
#pragma unroll
  for (int r = 0; r < 8; ++r) {
    float mfin = redmax16(ml[r]);
    float lfin = redsum16(sl[r] * __expf(ml[r] - mfin));
    lo[r] = mfin + __logf(lfin);
  }

  // ---------------- Pass 2: attn out + context = P @ V --------------------
  v8f ctx0 = {}, ctx1 = {}, ctx2 = {}, ctx3 = {};
  {
    {
      float4 kv = *(const float4*)(Kb + e0);
      store_bf16x4(&Klds[0][key0][dim0], kv);
      kv = *(const float4*)(Kb + e1);
      store_bf16x4(&Klds[0][key1][dim1], kv);
      float4 vv = *(const float4*)(Vb + e0);
      Vtlds[0][dim0 + 0][key0] = (__bf16)vv.x;
      Vtlds[0][dim0 + 1][key0] = (__bf16)vv.y;
      Vtlds[0][dim0 + 2][key0] = (__bf16)vv.z;
      Vtlds[0][dim0 + 3][key0] = (__bf16)vv.w;
      vv = *(const float4*)(Vb + e1);
      Vtlds[0][dim1 + 0][key1] = (__bf16)vv.x;
      Vtlds[0][dim1 + 1][key1] = (__bf16)vv.y;
      Vtlds[0][dim1 + 2][key1] = (__bf16)vv.z;
      Vtlds[0][dim1 + 3][key1] = (__bf16)vv.w;
    }
    __syncthreads();

    for (int kbi = 0; kbi < kbCount; ++kbi) {
      const bool haveNext = (kbi + 1 < kbCount);
      float4 kr0, kr1, vr0, vr1;
      if (haveNext) {                      // issue next block's loads early
        const float* ksrc = Kb + (size_t)(kbi + 1) * 32 * DKc;
        const float* vsrc = Vb + (size_t)(kbi + 1) * 32 * DKc;
        kr0 = *(const float4*)(ksrc + e0);
        kr1 = *(const float4*)(ksrc + e1);
        vr0 = *(const float4*)(vsrc + e0);
        vr1 = *(const float4*)(vsrc + e1);
      }
      if (kbi <= kbNeed) {                 // wave-uniform
        const int cur = kbi & 1;
        const __bf16 (*kcur)[68] = Klds[cur];
        const __bf16 (*vcur)[36] = Vtlds[cur];
        const int kb = kbi * 32;

        v8f s0 = {}, s1 = {};
        {
          v16bf b = load_k_frag(kcur, 0, lane, 0);
          s0 = wmma_acc(qa0, b, s0);
          b = load_k_frag(kcur, 0, lane, 1);
          s0 = wmma_acc(qa1, b, s0);
          b = load_k_frag(kcur, 1, lane, 0);
          s1 = wmma_acc(qa0, b, s1);
          b = load_k_frag(kcur, 1, lane, 1);
          s1 = wmma_acc(qa1, b, s1);
        }
        const int col0 = kb + ln, col1 = kb + 16 + ln;
        const int rowBase = q0w + (hi ? 8 : 0);
        const bool diag = (kbi == kbNeed);
#pragma unroll
        for (int r = 0; r < 8; ++r) {
          const int row = rowBase + r;
          float sv0 = s0[r], sv1 = s1[r];
          if (diag) {                      // wave-uniform predicate
            sv0 = (col0 > row) ? -1.0e9f : sv0;
            sv1 = (col1 > row) ? -1.0e9f : sv1;
          }
          float p0 = __expf(sv0 - lo[r]);  // masked -> exact 0
          float p1 = __expf(sv1 - lo[r]);
          __builtin_nontemporal_store(p0, &attnb[(size_t)row * Sc + col0]);
          __builtin_nontemporal_store(p1, &attnb[(size_t)row * Sc + col1]);
          const int rloc = r + (hi ? 8 : 0);
          Plds[wv][rloc][ln]      = (__bf16)p0;    // C-layout -> row-major LDS
          Plds[wv][rloc][16 + ln] = (__bf16)p1;
        }
        // Wave-local LDS turnaround (no cross-wave sharing of Plds[wv])
        __builtin_amdgcn_wave_barrier();
        asm volatile("s_wait_dscnt 0" ::: "memory");

        v16bf pa;                          // A-fragment of P (16x32)
#pragma unroll
        for (int e = 0; e < 16; ++e) {
          int k = kpat(e >> 1, hi) + (e & 1);
          pa[e] = Plds[wv][ln][k];
        }
#pragma unroll
        for (int nt = 0; nt < 4; ++nt) {   // context dk tiles
          v16bf vb;
#pragma unroll
          for (int e = 0; e < 16; ++e) {
            int k = kpat(e >> 1, hi) + (e & 1);
            vb[e] = vcur[nt * 16 + ln][k];
          }
          if      (nt == 0) ctx0 = wmma_acc(pa, vb, ctx0);
          else if (nt == 1) ctx1 = wmma_acc(pa, vb, ctx1);
          else if (nt == 2) ctx2 = wmma_acc(pa, vb, ctx2);
          else              ctx3 = wmma_acc(pa, vb, ctx3);
        }
      }
      if (haveNext) {                      // convert + store into other buffer
        const int nb = (kbi + 1) & 1;
        store_bf16x4(&Klds[nb][key0][dim0], kr0);
        store_bf16x4(&Klds[nb][key1][dim1], kr1);
        Vtlds[nb][dim0 + 0][key0] = (__bf16)vr0.x;
        Vtlds[nb][dim0 + 1][key0] = (__bf16)vr0.y;
        Vtlds[nb][dim0 + 2][key0] = (__bf16)vr0.z;
        Vtlds[nb][dim0 + 3][key0] = (__bf16)vr0.w;
        Vtlds[nb][dim1 + 0][key1] = (__bf16)vr1.x;
        Vtlds[nb][dim1 + 1][key1] = (__bf16)vr1.y;
        Vtlds[nb][dim1 + 2][key1] = (__bf16)vr1.z;
        Vtlds[nb][dim1 + 3][key1] = (__bf16)vr1.w;
      }
      __syncthreads();
    }
  }

  // ---------------- Write context (streaming) ------------------------------
  {
    const int rowBase = q0w + (hi ? 8 : 0);
#pragma unroll
    for (int r = 0; r < 8; ++r) {
      float* o = ctxb + (size_t)(rowBase + r) * DKc + ln;
      __builtin_nontemporal_store(ctx0[r], o);
      __builtin_nontemporal_store(ctx1[r], o + 16);
      __builtin_nontemporal_store(ctx2[r], o + 32);
      __builtin_nontemporal_store(ctx3[r], o + 48);
    }
  }

  // ---------------- Zero fully-masked attn tail (b128 NT stores) ----------
  {
    const int colZ = (kbNeed + 1) * 32;     // first never-computed column
    const v4f z = {0.f, 0.f, 0.f, 0.f};
    for (int r = 0; r < 16; ++r) {
      const int row = q0w + r;
      for (int c = colZ + lane * 4; c < Sc; c += 128) {
        __builtin_nontemporal_store(z, (v4f*)&attnb[(size_t)row * Sc + c]);
      }
    }
  }
}

extern "C" void kernel_launch(void* const* d_in, const int* in_sizes, int n_in,
                              void* d_out, int out_size, void* d_ws, size_t ws_size,
                              hipStream_t stream) {
  const float* Q = (const float*)d_in[0];
  const float* K = (const float*)d_in[1];
  const float* V = (const float*)d_in[2];
  // d_in[3] is the causal mask; recomputed analytically in-kernel.
  float* ctx  = (float*)d_out;
  float* attn = ctx + (size_t)Bc * Hc * Sc * DKc;   // tuple output: ctx then attn

  dim3 grid(Sc / 128, Bc * Hc);
  fa_fwd_kernel<<<grid, 256, 0, stream>>>(Q, K, V, ctx, attn);
}
